// GAT16Model_6124623364724
// MI455X (gfx1250) — compile-verified, matrix operand order checked by reading, not run on hardware
//
#include <hip/hip_runtime.h>
#include <hip/hip_bf16.h>

#define CH 64

typedef __bf16 bvec8  __attribute__((ext_vector_type(8)));
typedef __bf16 bvec16 __attribute__((ext_vector_type(16)));
typedef float  fvec8  __attribute__((ext_vector_type(8)));

__device__ __forceinline__ float lrelu(float v){ return v > 0.f ? v : 0.2f * v; }

// float atomic max via signed/unsigned int ordering trick
__device__ __forceinline__ void atomicMaxF(float* addr, float val){
  if (val >= 0.f) atomicMax((int*)addr, __float_as_int(val));
  else            atomicMin((unsigned int*)addr, (unsigned int)__float_as_int(val));
}

// edges [0,E) are real edges; [E, E+N) are self loops with attr = loop_attr[node]
__device__ __forceinline__ void edge_get(int e, int E, const int* src, const int* dst,
                                         const float* w, const float* lattr,
                                         int& s, int& d, float& wt){
  if (e < E){ s = src[e]; d = dst[e]; wt = w[e]; }
  else { s = e - E; d = s; wt = lattr[s]; }
}

__device__ __forceinline__ bvec16 load_frag16(const __bf16* p){
  // two contiguous 16B loads -> one 32B fragment (elements 0..7 = p[0..7], 8..15 = p[16..23])
  bvec8 lo = *(const bvec8*)(p);
  bvec8 hi = *(const bvec8*)(p + 16);
  return __builtin_shufflevector(lo, hi, 0,1,2,3,4,5,6,7,8,9,10,11,12,13,14,15);
}

// ---------------- setup kernels ----------------

__global__ void k_zero2(float* a, float* b, int n){
  int i = blockIdx.x * blockDim.x + threadIdx.x;
  if (i < n){ a[i] = 0.f; b[i] = 0.f; }
}

__global__ void k_deg(const int* __restrict__ dst, const float* __restrict__ w,
                      float* cnt, float* wsum, int E){
  int e = blockIdx.x * blockDim.x + threadIdx.x;
  if (e < E){
    atomicAdd(cnt + dst[e], 1.f);
    atomicAdd(wsum + dst[e], w[e]);
  }
}

__global__ void k_lattr(const float* cnt, const float* wsum, float* lattr, int N){
  int i = blockIdx.x * blockDim.x + threadIdx.x;
  if (i < N) lattr[i] = wsum[i] / fmaxf(cnt[i], 1.f);
}

__global__ void k_f2b(const float* __restrict__ x, __hip_bfloat16* __restrict__ xb, int n){
  int i = blockIdx.x * blockDim.x + threadIdx.x;
  if (i < n) xb[i] = __float2bfloat16(x[i]);
}

// convert AND transpose weights: out[c*CH + k] = W[k*CH + c]
// so a WMMA B-column fragment is two contiguous 16B loads (same pattern as A rows).
__global__ void k_w2bT(const float* __restrict__ Wl, const float* __restrict__ Wr,
                       __hip_bfloat16* __restrict__ wlt, __hip_bfloat16* __restrict__ wrt){
  int i = blockIdx.x * blockDim.x + threadIdx.x;
  if (i < CH * CH){
    const int k = i >> 6, c = i & (CH - 1);
    wlt[c * CH + k] = __float2bfloat16(Wl[i]);
    wrt[c * CH + k] = __float2bfloat16(Wr[i]);
  }
}

// ---------------- WMMA dual GEMM: xl = x@Wl+bl, xr = x@Wr+br ----------------
// One wave = one 16-row M tile, computing all 4 N-tiles for BOTH Wl and Wr
// (8 accumulator tiles, A fragment reused 8x). Block = 8 waves = 128 rows.
// K=64 -> 2 k-steps -> 16 v_wmma_f32_16x16x32_bf16 per wave.
__global__ __launch_bounds__(256) void k_gemm(
    const __hip_bfloat16* __restrict__ Xh,
    const __hip_bfloat16* __restrict__ Wlt,   // transposed bf16 [CH x CH]
    const __hip_bfloat16* __restrict__ Wrt,   // transposed bf16 [CH x CH]
    const float* __restrict__ bl, const float* __restrict__ br,
    float* __restrict__ xl, float* __restrict__ xr, int N)
{
  const __bf16* X = reinterpret_cast<const __bf16*>(Xh);
  const __bf16* WT0 = reinterpret_cast<const __bf16*>(Wlt);
  const __bf16* WT1 = reinterpret_cast<const __bf16*>(Wrt);

  const int wave = threadIdx.x >> 5;
  const int lane = threadIdx.x & 31;
  const int tile = blockIdx.x * 8 + wave;           // 16-row M tile
  if (tile * 16 >= N) return;
  const bool tilefull = (tile * 16 + 16 <= N);

  const int colw  = lane & 15;                      // lane's column within an N tile
  const int khalf = (lane < 16) ? 0 : 8;            // ISA 16-bit A/B K split per half-wave
  int row = tile * 16 + colw;                       // lane's A row
  if (row >= N) row = N - 1;

  fvec8 acc[2][4];
#pragma unroll
  for (int nt = 0; nt < 4; ++nt){
    const float bvl = bl[nt * 16 + colw];
    const float bvr = br[nt * 16 + colw];
#pragma unroll
    for (int v = 0; v < 8; ++v){ acc[0][nt][v] = bvl; acc[1][nt][v] = bvr; }
  }

#pragma unroll
  for (int ks = 0; ks < 2; ++ks){
    const int k0 = ks * 32 + khalf;
    const bvec16 a = load_frag16(X + row * CH + k0);
#pragma unroll
    for (int nt = 0; nt < 4; ++nt){
      const int c = nt * 16 + colw;
      const bvec16 b0 = load_frag16(WT0 + c * CH + k0);
      acc[0][nt] = __builtin_amdgcn_wmma_f32_16x16x32_bf16(
          false, a, false, b0, (short)0, acc[0][nt], false, false);
      const bvec16 b1 = load_frag16(WT1 + c * CH + k0);
      acc[1][nt] = __builtin_amdgcn_wmma_f32_16x16x32_bf16(
          false, a, false, b1, (short)0, acc[1][nt], false, false);
    }
  }

  const int rbase = tile * 16 + ((lane < 16) ? 0 : 8);
  if (tilefull){
#pragma unroll
    for (int v = 0; v < 8; ++v){
      const int r = rbase + v;
#pragma unroll
      for (int nt = 0; nt < 4; ++nt){
        xl[r * CH + nt * 16 + colw] = acc[0][nt][v];
        xr[r * CH + nt * 16 + colw] = acc[1][nt][v];
      }
    }
  } else {
#pragma unroll
    for (int v = 0; v < 8; ++v){
      const int r = rbase + v;
      if (r < N){
#pragma unroll
        for (int nt = 0; nt < 4; ++nt){
          xl[r * CH + nt * 16 + colw] = acc[0][nt][v];
          xr[r * CH + nt * 16 + colw] = acc[1][nt][v];
        }
      }
    }
  }
}

// ---------------- per-layer node init: out=bias, m=-inf, ssum=0 ----------------
__global__ void k_nodeinit(float* __restrict__ out, const float* __restrict__ bo,
                           float* m, float* ssum, int N){
  int i = blockIdx.x * blockDim.x + threadIdx.x;
  if (i >= N * CH) return;
  const int c = i & (CH - 1);
  out[i] = bo[c];
  if (c == 0){ m[i >> 6] = -__builtin_huge_valf(); ssum[i >> 6] = 0.f; }
}

// ---------------- edge logit + segment max ----------------
__global__ void k_logit(const float* __restrict__ xl, const float* __restrict__ xr,
                        const float* __restrict__ We, const float* __restrict__ att,
                        const int* __restrict__ src, const int* __restrict__ dst,
                        const float* __restrict__ ew, const float* __restrict__ lattr,
                        float* __restrict__ logitz, float* m, int E, int Ea)
{
  int e = blockIdx.x * blockDim.x + threadIdx.x;
  if (e >= Ea) return;
  int s, d; float wt;
  edge_get(e, E, src, dst, ew, lattr, s, d, wt);
  const float4* pl = (const float4*)(xl + s * CH);
  const float4* pr = (const float4*)(xr + d * CH);
  const float4* pw = (const float4*)We;
  const float4* pa = (const float4*)att;
  float acc = 0.f;
#pragma unroll 4
  for (int j = 0; j < CH / 4; ++j){
    float4 a = pl[j], b = pr[j], w4 = pw[j], t = pa[j];
    acc += lrelu(a.x + b.x + wt * w4.x) * t.x;
    acc += lrelu(a.y + b.y + wt * w4.y) * t.y;
    acc += lrelu(a.z + b.z + wt * w4.z) * t.z;
    acc += lrelu(a.w + b.w + wt * w4.w) * t.w;
  }
  logitz[e] = acc;
  atomicMaxF(m + d, acc);
}

// ---------------- exp(logit - m[dst]) + segment sum ----------------
__global__ void k_expsum(const int* __restrict__ dst, float* __restrict__ logitz,
                         const float* __restrict__ m, float* ssum, int E, int Ea){
  int e = blockIdx.x * blockDim.x + threadIdx.x;
  if (e >= Ea) return;
  const int d = (e < E) ? dst[e] : (e - E);
  const float z = expf(logitz[e] - m[d]);
  logitz[e] = z;
  atomicAdd(ssum + d, z);
}

// ---------------- scatter alpha * xl[src] into out[dst] (wave per edge) ----------------
__global__ void k_scatter(const int* __restrict__ src, const int* __restrict__ dst,
                          const float* __restrict__ logitz, const float* __restrict__ ssum,
                          const float* __restrict__ xl, float* __restrict__ out, int E, int Ea){
  int gid = blockIdx.x * blockDim.x + threadIdx.x;
  int e = gid >> 5;
  if (e >= Ea) return;
  int lane = gid & 31;
  int s, d;
  if (e < E){ s = src[e]; d = dst[e]; } else { s = d = e - E; }
  const float alpha = logitz[e] / ssum[d];
  atomicAdd(out + d * CH + lane,      alpha * xl[s * CH + lane]);
  atomicAdd(out + d * CH + lane + 32, alpha * xl[s * CH + lane + 32]);
}

// ---------------- final layer (CH -> 1) scalar path ----------------
__global__ void k_dot8(const float* __restrict__ x,
                       const float* __restrict__ Wl8, const float* __restrict__ bl8,
                       const float* __restrict__ Wr8, const float* __restrict__ br8,
                       float* xl8, float* xr8, int N){
  int n = blockIdx.x * blockDim.x + threadIdx.x;
  if (n >= N) return;
  const float4* px = (const float4*)(x + n * CH);
  const float4* pl = (const float4*)Wl8;
  const float4* pr = (const float4*)Wr8;
  float al = 0.f, ar = 0.f;
#pragma unroll 4
  for (int j = 0; j < CH / 4; ++j){
    float4 v = px[j], a = pl[j], b = pr[j];
    al += v.x * a.x + v.y * a.y + v.z * a.z + v.w * a.w;
    ar += v.x * b.x + v.y * b.y + v.z * b.z + v.w * b.w;
  }
  xl8[n] = al + bl8[0];
  xr8[n] = ar + br8[0];
}

__global__ void k_init8(float* out8, const float* bo8, float* m, float* ssum, int N){
  int n = blockIdx.x * blockDim.x + threadIdx.x;
  if (n < N){ out8[n] = bo8[0]; m[n] = -__builtin_huge_valf(); ssum[n] = 0.f; }
}

__global__ void k_logit8(const float* __restrict__ xl8, const float* __restrict__ xr8,
                         const float* __restrict__ We8, const float* __restrict__ att8,
                         const int* __restrict__ src, const int* __restrict__ dst,
                         const float* __restrict__ ew, const float* __restrict__ lattr,
                         float* __restrict__ logitz, float* m, int E, int Ea){
  int e = blockIdx.x * blockDim.x + threadIdx.x;
  if (e >= Ea) return;
  int s, d; float wt;
  edge_get(e, E, src, dst, ew, lattr, s, d, wt);
  const float lg = lrelu(xl8[s] + xr8[d] + wt * We8[0]) * att8[0];
  logitz[e] = lg;
  atomicMaxF(m + d, lg);
}

__global__ void k_scatter8(const int* __restrict__ src, const int* __restrict__ dst,
                           const float* __restrict__ logitz, const float* __restrict__ ssum,
                           const float* __restrict__ xl8, float* out8, int E, int Ea){
  int e = blockIdx.x * blockDim.x + threadIdx.x;
  if (e >= Ea) return;
  int s, d;
  if (e < E){ s = src[e]; d = dst[e]; } else { s = d = e - E; }
  atomicAdd(out8 + d, (logitz[e] / ssum[d]) * xl8[s]);
}

__global__ void k_zero1(float* o){
  if (blockIdx.x == 0 && threadIdx.x == 0) o[0] = 0.f;
}

__global__ void k_mean(const float* __restrict__ out8, float* o, int N){
  int n = blockIdx.x * blockDim.x + threadIdx.x;
  if (n < N) atomicAdd(o, out8[n] * (1.0f / (float)N));
}

// ---------------- driver ----------------

extern "C" void kernel_launch(void* const* d_in, const int* in_sizes, int n_in,
                              void* d_out, int out_size, void* d_ws, size_t ws_size,
                              hipStream_t stream){
  const float* features = (const float*)d_in[0];
  const int*   esrc     = (const int*)d_in[1];
  const int*   edst     = (const int*)d_in[2];
  const float* ew       = (const float*)d_in[3];
  const float* P[21];
  for (int i = 0; i < 21; ++i) P[i] = (const float*)d_in[4 + i];

  const int N  = in_sizes[0] / CH;
  const int E  = in_sizes[1];
  const int Ea = E + N;

  size_t off = 0;
  auto alloc = [&](size_t bytes) -> void* {
    void* p = (char*)d_ws + off;
    off += (bytes + 255) & ~(size_t)255;
    return p;
  };
  float* xA   = (float*)alloc((size_t)N * CH * 4);
  float* xB   = (float*)alloc((size_t)N * CH * 4);
  float* xl   = (float*)alloc((size_t)N * CH * 4);
  float* xr   = (float*)alloc((size_t)N * CH * 4);
  __hip_bfloat16* xb  = (__hip_bfloat16*)alloc((size_t)N * CH * 2);
  __hip_bfloat16* wlt = (__hip_bfloat16*)alloc(CH * CH * 2);
  __hip_bfloat16* wrt = (__hip_bfloat16*)alloc(CH * CH * 2);
  float* logitz = (float*)alloc((size_t)Ea * 4);
  float* m    = (float*)alloc((size_t)N * 4);
  float* ssum = (float*)alloc((size_t)N * 4);
  float* cnt  = (float*)alloc((size_t)N * 4);
  float* wsum = (float*)alloc((size_t)N * 4);
  float* lattr= (float*)alloc((size_t)N * 4);
  float* xl8  = (float*)alloc((size_t)N * 4);
  float* xr8  = (float*)alloc((size_t)N * 4);
  float* out8 = (float*)alloc((size_t)N * 4);
  (void)ws_size; (void)n_in; (void)out_size;

  const int B = 256;
  auto g = [&](long long n){ return (int)((n + B - 1) / B); };

  // self-loop attrs: mean of incoming edge weights
  k_zero2<<<g(N), B, 0, stream>>>(cnt, wsum, N);
  k_deg  <<<g(E), B, 0, stream>>>(edst, ew, cnt, wsum, E);
  k_lattr<<<g(N), B, 0, stream>>>(cnt, wsum, lattr, N);

  const float* xin = features;
  float* xout = xA;
  const int gemmBlocks = (N + 127) / 128;   // 8 waves/block, 16 rows/wave

  for (int l = 0; l < 7; ++l){
    const float *Wl, *bl, *Wr, *br, *We, *att, *bo;
    if (l == 0){
      Wl = P[0]; bl = P[1]; Wr = P[2]; br = P[3]; We = P[4]; att = P[5]; bo = P[6];
    } else {
      const int i = l - 1;
      Wl = P[7]  + (size_t)i * CH * CH; bl  = P[8]  + i * CH;
      Wr = P[9]  + (size_t)i * CH * CH; br  = P[10] + i * CH;
      We = P[11] + i * CH;              att = P[12] + i * CH;
      bo = P[13] + i * CH;
    }
    k_w2bT<<<g(CH * CH), B, 0, stream>>>(Wl, Wr, wlt, wrt);
    k_f2b<<<g((long long)N * CH), B, 0, stream>>>(xin, xb, N * CH);
    k_gemm<<<gemmBlocks, 256, 0, stream>>>(xb, wlt, wrt, bl, br, xl, xr, N);
    k_nodeinit<<<g((long long)N * CH), B, 0, stream>>>(xout, bo, m, ssum, N);
    k_logit<<<g(Ea), B, 0, stream>>>(xl, xr, We, att, esrc, edst, ew, lattr, logitz, m, E, Ea);
    k_expsum<<<g(Ea), B, 0, stream>>>(edst, logitz, m, ssum, E, Ea);
    k_scatter<<<g((long long)Ea * 32), B, 0, stream>>>(esrc, edst, logitz, ssum, xl, xout, E, Ea);
    xin = xout;
    xout = (xout == xA) ? xB : xA;
  }

  // layer 8: CH -> 1
  k_dot8<<<g(N), B, 0, stream>>>(xin, P[14], P[15], P[16], P[17], xl8, xr8, N);
  k_init8<<<g(N), B, 0, stream>>>(out8, P[20], m, ssum, N);
  k_logit8<<<g(Ea), B, 0, stream>>>(xl8, xr8, P[18], P[19], esrc, edst, ew, lattr, logitz, m, E, Ea);
  k_expsum<<<g(Ea), B, 0, stream>>>(edst, logitz, m, ssum, E, Ea);
  k_scatter8<<<g(Ea), B, 0, stream>>>(esrc, edst, logitz, ssum, xl8, out8, E, Ea);
  k_zero1<<<1, 32, 0, stream>>>((float*)d_out);
  k_mean<<<g(N), B, 0, stream>>>(out8, (float*)d_out, N);
}
